// GAT_9732395892850
// MI455X (gfx1250) — compile-verified, matrix operand order checked by reading, not run on hardware
//
#include <hip/hip_runtime.h>
#include <math.h>
#include <cstddef>

// ---------------- problem constants (match reference) ----------------
#define NNODES   20000
#define NEDGES   320000
#define ETOT     (NEDGES + NNODES)     // self-loops appended
#define IN_CH    128
#define HID_CH   32
#define HEADS    8
#define H1C      (HEADS * HID_CH)      // 256
#define OUT_CH   64
#define NEG_SLOPE 0.2f

typedef float  v2f   __attribute__((ext_vector_type(2)));
typedef float  v8f   __attribute__((ext_vector_type(8)));
typedef __bf16 v16bf __attribute__((ext_vector_type(16)));

// ---------------------------------------------------------------------
// WMMA GEMM: C[M,N] = A[M,K] * B[K,N], fp32 row-major, M%16==K%32==N%16==0.
// grid = (M/16, N/16), block = 32 threads (one wave32; EXEC all ones).
// Prefers V_WMMA_F32_16X16X4_F32 (full fp32); falls back to bf16 WMMA.
// ---------------------------------------------------------------------
__global__ void wmma_gemm(const float* __restrict__ A, const float* __restrict__ B,
                          float* __restrict__ C, int M, int K, int N) {
  const int lane = threadIdx.x;
  const int row  = lane & 15;     // M (for A) / N (for B) position of this lane
  const int kh   = lane >> 4;     // which K-half this lane's VGPRs hold
  const int tm   = blockIdx.x;
  const int tn   = blockIdx.y;
  v8f acc = {0.f, 0.f, 0.f, 0.f, 0.f, 0.f, 0.f, 0.f};

#if __has_builtin(__builtin_amdgcn_wmma_f32_16x16x4_f32)
  // fp32 path: A frag 16x4 (2 VGPR), B frag 4x16 (2 VGPR), C/D 16x16 (8 VGPR)
  const float* Arow = A + (size_t)(tm * 16 + row) * K;
  const float* Bcol = B + (size_t)(tn * 16 + row);
  for (int k = 0; k < K; k += 4) {
    v2f a, b;
    a.x = Arow[k + kh * 2 + 0];                 // A[m][k'],   k' = kh*2 + r
    a.y = Arow[k + kh * 2 + 1];
    b.x = Bcol[(size_t)(k + kh * 2 + 0) * N];   // B[k'][n]
    b.y = Bcol[(size_t)(k + kh * 2 + 1) * N];
    acc = __builtin_amdgcn_wmma_f32_16x16x4_f32(false, a, false, b,
                                                (short)0, acc, false, false);
  }
#elif __has_builtin(__builtin_amdgcn_wmma_f32_16x16x32_bf16)
  // bf16 fallback (codegen-confirmed): convert fp32->bf16 in flight.
  // 16-bit A 16x32 layout: lane -> m=lane&15; vgpr v, half p:
  //   k = (v>=4 ? 16 : 0) + kh*8 + (v&3)*2 + p    (B is the K-major mirror)
  for (int k0 = 0; k0 < K; k0 += 32) {
    v16bf a, b;
#pragma unroll
    for (int i = 0; i < 16; ++i) {
      int vg = i >> 1, p = i & 1;
      int k  = k0 + ((vg & 4) ? 16 : 0) + kh * 8 + (vg & 3) * 2 + p;
      a[i] = (__bf16)A[(size_t)(tm * 16 + row) * K + k];
      b[i] = (__bf16)B[(size_t)k * N + tn * 16 + row];
    }
    acc = __builtin_amdgcn_wmma_f32_16x16x32_bf16(false, a, false, b,
                                                  (short)0, acc, false, false);
  }
#endif

  // D 16x16 layout: VGPR r -> m = r + kh*8, n = row
  float* Crow = C + ((size_t)tm * 16 + kh * 8) * N + tn * 16 + row;
#pragma unroll
  for (int r = 0; r < 8; ++r) Crow[(size_t)r * N] = acc[r];
}

// ---------------------------------------------------------------------
// helpers
// ---------------------------------------------------------------------
__global__ void fill_f32(float* __restrict__ p, float v, int n) {
  int i = blockIdx.x * blockDim.x + threadIdx.x;
  if (i < n) p[i] = v;
}

__device__ __forceinline__ void atomicMaxF(float* addr, float val) {
  // ordered-int trick: valid for all finite/-inf values, init must be -inf
  if (val >= 0.f) atomicMax((int*)addr, __float_as_int(val));
  else            atomicMin((unsigned int*)addr, __float_as_uint(val));
}

// per-(node,head): alpha_src/alpha_dst = <h[n,h,:], a_src/dst[h,:]>
__global__ void alpha_kernel(const float* __restrict__ h,
                             const float* __restrict__ a_src,
                             const float* __restrict__ a_dst,
                             float* __restrict__ as, float* __restrict__ ad,
                             int N, int H, int C) {
  int i = blockIdx.x * blockDim.x + threadIdx.x;     // i = n*H + head
  if (i >= N * H) return;
  int hd = i % H;
  const float* hp = h + (size_t)i * C;
  float s = 0.f, d = 0.f;
  for (int c = 0; c < C; ++c) {
    float v = hp[c];
    s += v * a_src[hd * C + c];
    d += v * a_dst[hd * C + c];
  }
  as[i] = s;
  ad[i] = d;
}

// edge pass 1: e = leaky_relu(as[src]+ad[dst]); segment atomic max per dst
__global__ void edge_max_kernel(const int* __restrict__ ei,
                                const float* __restrict__ as,
                                const float* __restrict__ ad,
                                float* __restrict__ e, float* __restrict__ m,
                                int Eorig, int Etot, int H) {
  int i = blockIdx.x * blockDim.x + threadIdx.x;     // i = eid*H + head
  if (i >= Etot * H) return;
  int eid = i / H, hd = i % H;
  int s, d;
  if (eid < Eorig) { s = ei[eid]; d = ei[Eorig + eid]; }
  else             { s = d = eid - Eorig; }           // self loop
  float v = as[s * H + hd] + ad[d * H + hd];
  v = (v > 0.f) ? v : NEG_SLOPE * v;
  e[i] = v;
  atomicMaxF(&m[d * H + hd], v);
}

// edge pass 2a: ex = exp(e - m[dst]) (stored back), denom[dst] += ex
__global__ void edge_expsum_kernel(const int* __restrict__ ei,
                                   float* __restrict__ e,
                                   const float* __restrict__ m,
                                   float* __restrict__ den,
                                   int Eorig, int Etot, int H) {
  int i = blockIdx.x * blockDim.x + threadIdx.x;
  if (i >= Etot * H) return;
  int eid = i / H, hd = i % H;
  int d = (eid < Eorig) ? ei[Eorig + eid] : (eid - Eorig);
  float ex = __expf(e[i] - m[d * H + hd]);
  e[i] = ex;
  atomicAdd(&den[d * H + hd], ex);
}

// edge pass 2b: acc[dst, ch] += ex * h[src, ch]   (coalesced over channels)
__global__ void edge_scatter_kernel(const int* __restrict__ ei,
                                    const float* __restrict__ e,
                                    const float* __restrict__ h,
                                    float* __restrict__ acc,
                                    int Eorig, int Etot, int H, int C) {
  long long i = (long long)blockIdx.x * blockDim.x + threadIdx.x;
  int HC = H * C;
  if (i >= (long long)Etot * HC) return;
  int eid = (int)(i / HC);
  int ch  = (int)(i % HC);
  int hd  = ch / C;
  int s, d;
  if (eid < Eorig) { s = ei[eid]; d = ei[Eorig + eid]; }
  else             { s = d = eid - Eorig; }
  atomicAdd(&acc[(size_t)d * HC + ch],
            e[(size_t)eid * H + hd] * h[(size_t)s * HC + ch]);
}

// per-(node,channel): out = acc/(denom+eps) + bias, optional ELU
__global__ void finalize_kernel(const float* __restrict__ acc,
                                const float* __restrict__ den,
                                const float* __restrict__ bias,
                                float* __restrict__ out,
                                int N, int H, int C, int do_elu) {
  int i = blockIdx.x * blockDim.x + threadIdx.x;
  int HC = H * C;
  if (i >= N * HC) return;
  int n = i / HC, ch = i % HC, hd = ch / C;
  float v = acc[i] / (den[n * H + hd] + 1e-16f) + bias[ch];
  if (do_elu) v = (v > 0.f) ? v : (__expf(v) - 1.f);
  out[i] = v;
}

// ---------------------------------------------------------------------
extern "C" void kernel_launch(void* const* d_in, const int* in_sizes, int n_in,
                              void* d_out, int out_size, void* d_ws, size_t ws_size,
                              hipStream_t stream) {
  const float* x      = (const float*)d_in[0];
  const int*   ei     = (const int*)  d_in[1];   // [2, NEDGES]
  const float* W1     = (const float*)d_in[2];   // [128,256]
  const float* a_src1 = (const float*)d_in[3];   // [8,32]
  const float* a_dst1 = (const float*)d_in[4];
  const float* b1     = (const float*)d_in[5];   // [256]
  const float* W2     = (const float*)d_in[6];   // [256,64]
  const float* a_src2 = (const float*)d_in[7];   // [1,64]
  const float* a_dst2 = (const float*)d_in[8];
  const float* b2     = (const float*)d_in[9];   // [64]
  float* out = (float*)d_out;                    // [20000,64]

  // workspace layout (floats) — ~54.4 MB total
  float* ws   = (float*)d_ws;
  float* h1   = ws;                              // 20000*256 (reused as ELU out)
  float* acc1 = h1   + (size_t)NNODES * H1C;     // 20000*256
  float* ebuf = acc1 + (size_t)NNODES * H1C;     // 340000*8
  float* as1  = ebuf + (size_t)ETOT * HEADS;     // 20000*8
  float* ad1  = as1  + (size_t)NNODES * HEADS;
  float* m1   = ad1  + (size_t)NNODES * HEADS;
  float* den1 = m1   + (size_t)NNODES * HEADS;
  // layer-2 scratch reuses acc1's region (free after finalize1)
  float* h2   = acc1;                            // 20000*64
  float* as2  = acc1 + (size_t)NNODES * OUT_CH;  // 20000 each
  float* ad2  = as2 + NNODES;
  float* m2   = ad2 + NNODES;
  float* den2 = m2  + NNODES;

  const int T = 256;
  const int NH1 = NNODES * HEADS;                // 160000
  const int EH1 = ETOT * HEADS;                  // 2.72M
  const long long SC1 = (long long)ETOT * H1C;   // 87.04M
  const long long SC2 = (long long)ETOT * OUT_CH;

  // -------- layer 1 --------
  fill_f32<<<(NH1 + T - 1) / T, T, 0, stream>>>(m1, -INFINITY, NH1);
  fill_f32<<<(NH1 + T - 1) / T, T, 0, stream>>>(den1, 0.f, NH1);
  fill_f32<<<(NNODES * H1C + T - 1) / T, T, 0, stream>>>(acc1, 0.f, NNODES * H1C);

  wmma_gemm<<<dim3(NNODES / 16, H1C / 16), 32, 0, stream>>>(x, W1, h1,
                                                            NNODES, IN_CH, H1C);
  alpha_kernel<<<(NH1 + T - 1) / T, T, 0, stream>>>(h1, a_src1, a_dst1, as1, ad1,
                                                    NNODES, HEADS, HID_CH);
  edge_max_kernel<<<(EH1 + T - 1) / T, T, 0, stream>>>(ei, as1, ad1, ebuf, m1,
                                                       NEDGES, ETOT, HEADS);
  edge_expsum_kernel<<<(EH1 + T - 1) / T, T, 0, stream>>>(ei, ebuf, m1, den1,
                                                          NEDGES, ETOT, HEADS);
  edge_scatter_kernel<<<(unsigned)((SC1 + T - 1) / T), T, 0, stream>>>(
      ei, ebuf, h1, acc1, NEDGES, ETOT, HEADS, HID_CH);
  finalize_kernel<<<(NNODES * H1C + T - 1) / T, T, 0, stream>>>(
      acc1, den1, b1, h1, NNODES, HEADS, HID_CH, /*elu=*/1);   // h1 <- ELU(gat1)

  // -------- layer 2 --------
  fill_f32<<<(NNODES + T - 1) / T, T, 0, stream>>>(m2, -INFINITY, NNODES);
  fill_f32<<<(NNODES + T - 1) / T, T, 0, stream>>>(den2, 0.f, NNODES);
  fill_f32<<<(NNODES * OUT_CH + T - 1) / T, T, 0, stream>>>(out, 0.f, NNODES * OUT_CH);

  wmma_gemm<<<dim3(NNODES / 16, OUT_CH / 16), 32, 0, stream>>>(h1, W2, h2,
                                                               NNODES, H1C, OUT_CH);
  alpha_kernel<<<(NNODES + T - 1) / T, T, 0, stream>>>(h2, a_src2, a_dst2, as2, ad2,
                                                       NNODES, 1, OUT_CH);
  edge_max_kernel<<<(ETOT + T - 1) / T, T, 0, stream>>>(ei, as2, ad2, ebuf, m2,
                                                        NEDGES, ETOT, 1);
  edge_expsum_kernel<<<(ETOT + T - 1) / T, T, 0, stream>>>(ei, ebuf, m2, den2,
                                                           NEDGES, ETOT, 1);
  edge_scatter_kernel<<<(unsigned)((SC2 + T - 1) / T), T, 0, stream>>>(
      ei, ebuf, h2, out, NEDGES, ETOT, 1, OUT_CH);
  finalize_kernel<<<(NNODES * OUT_CH + T - 1) / T, T, 0, stream>>>(
      out, den2, b2, out, NNODES, 1, OUT_CH, /*elu=*/0);
}
